// CapsNet_833223655942
// MI455X (gfx1250) — matrix operand order repaired and checked by reading, not compiled, not run on hardware
//
#include <hip/hip_runtime.h>
#include <math.h>

typedef __attribute__((ext_vector_type(16))) __bf16 v16bf;
typedef __attribute__((ext_vector_type(8)))  float  v8f;
typedef __attribute__((ext_vector_type(4)))  float  v4f;
typedef __attribute__((ext_vector_type(8)))  unsigned int v8u;
typedef __attribute__((ext_vector_type(4)))  unsigned int u32x4;
typedef __attribute__((ext_vector_type(8)))  int i32x8;
typedef __attribute__((ext_vector_type(4)))  int i32x4;

#define BATCH      4096
#define L1_M       (BATCH*42)     // 172032 rows for conv1 GEMM
#define L2_M       (BATCH*48)     // 196608 rows for primary-caps GEMM
#define HS_PER_B   2100           // 50*42 bf16 per batch
#define U_PER_B    3072           // 384*8 f32 per batch

// ---------------- workspace layout (bytes, 256-aligned) ----------------
#define OFF_HS   ((size_t)0)
#define OFF_U    ((size_t)17203456)                         // HS(17,203,200) + pad
#define OFF_BF1  ((size_t)67535104)
#define BF1_ELEM (36*4*32*16)                               // 73728 bf16 = 147456 B
#define OFF_BF2  ((size_t)67682560)
#define BF2_ELEM (4*4*32*16)                                // 8192 bf16 = 16384 B

// ======================================================================
// LDS helpers: workgroup-relative byte offset of an LDS pointer.
// ======================================================================
static __device__ __forceinline__ unsigned ldsOffset(const void* p) {
    return (unsigned)(unsigned long long)(const __attribute__((address_space(3))) char*)p;
}

// Stage a linear block of bf16 weights from global into LDS.
// Preferred path: Tensor Data Mover (async DMA, TENSORcnt), issued by wave 0.
static __device__ __forceinline__ void stage_to_lds(const __bf16* __restrict__ g,
                                                    __bf16* l, int elems,
                                                    int tid, int wid) {
#if __has_builtin(__builtin_amdgcn_tensor_load_to_lds)
    if (wid == 0) {
        unsigned bytes = (unsigned)elems * 2u;
        unsigned n8 = bytes >> 3;                       // 8-byte elements
        unsigned long long ga = (unsigned long long)g;
        unsigned ldsoff = ldsOffset(l);
        // D# group 0: count=1 | lds_addr | global_addr | type=2
        u32x4 g0;
        g0[0] = 1u;
        g0[1] = ldsoff;
        g0[2] = (unsigned)ga;
        g0[3] = (unsigned)(ga >> 32) | (2u << 30);
        // D# group 1: data_size=8B 1-D tensor: dim0=tile0=n8, dim1=tile1=1
        i32x8 g1;
        g1[0] = (int)(3u << 16);                        // data_size=3 (8B)
        g1[1] = (int)((n8 & 0xFFFFu) << 16);            // tensor_dim0[15:0]
        g1[2] = (int)((n8 >> 16) | (1u << 16));         // tensor_dim0[31:16] | tensor_dim1=1
        g1[3] = (int)((n8 & 0xFFFFu) << 16);            // tile_dim0 = n8
        g1[4] = 1;                                      // tile_dim1=1, tile_dim2=0
        g1[5] = (int)n8;                                // tensor_dim0_stride[31:0]
        g1[6] = (int)((n8 & 0xFFFFu) << 16);            // stride0 hi=0 | stride1[15:0]
        g1[7] = (int)(n8 >> 16);                        // stride1 hi
        i32x4 g2 = {0, 0, 0, 0};
        i32x4 g3 = {0, 0, 0, 0};
#if defined(__clang_major__) && (__clang_major__ >= 23)
        i32x8 g4 = {0, 0, 0, 0, 0, 0, 0, 0};
        __builtin_amdgcn_tensor_load_to_lds(g0, g1, g2, g3, g4, 0);
#else
        __builtin_amdgcn_tensor_load_to_lds(g0, g1, g2, g3, 0);
#endif
        __builtin_amdgcn_s_wait_tensorcnt(0);
    }
#else
    {
        const unsigned long long* gs = (const unsigned long long*)g;
        unsigned long long* ls = (unsigned long long*)l;
        for (int i = tid; i < elems / 4; i += 256) ls[i] = gs[i];
    }
#endif
    __syncthreads();
}

// ======================================================================
// Weight prep: swizzle conv1 weights into fragment-ready B layout
// Bf1[((ks*4+nt)*32+lane)*16+e] = Wmat[K][N], K=ks*32+(lane>=16?8:0)+(e&7)+(e>=8?16:0),
// N=nt*16+(lane&15); Wmat[j][o] = conv1_w[o,ci,kk] with j = kk*128+ci.
// ======================================================================
__global__ __launch_bounds__(256)
void prep_w1(const float* __restrict__ w, __bf16* __restrict__ Bf) {
    int idx = blockIdx.x * 256 + threadIdx.x;          // < 73728
    int e    = idx & 15;
    int lane = (idx >> 4) & 31;
    int nt   = (idx >> 9) & 3;
    int ks   = idx >> 11;                              // 0..35
    int K = ks * 32 + ((lane & 16) ? 8 : 0) + (e & 7) + ((e & 8) ? 16 : 0);      // 0..1151
    int n = nt * 16 + (lane & 15);                     // 0..63
    float v = 0.0f;
    if (n < 50) {
        int kk = K >> 7;          // row offset within patch (0..8)
        int ci = K & 127;         // input channel
        v = w[n * 1152 + ci * 9 + kk];
    }
    Bf[idx] = (__bf16)v;
}

// pconv weights: j = kk*42 + c (patch offset into hs rows), padded K->128
__global__ __launch_bounds__(256)
void prep_w2(const float* __restrict__ w, __bf16* __restrict__ Bf) {
    int idx = blockIdx.x * 256 + threadIdx.x;          // < 8192
    int e    = idx & 15;
    int lane = (idx >> 4) & 31;
    int nt   = (idx >> 9) & 3;
    int ks   = idx >> 11;                              // 0..3
    int K = ks * 32 + ((lane & 16) ? 8 : 0) + (e & 7) + ((e & 8) ? 16 : 0);      // 0..127
    int n = nt * 16 + (lane & 15);                     // 0..63
    float v = 0.0f;
    if (K < 126) {
        int kk = K / 42;
        int c  = K % 42;
        v = w[n * 126 + c * 3 + kk];
    }
    Bf[idx] = (__bf16)v;
}

// ======================================================================
// GEMM1: conv1 + bias + ReLU.  out stored transposed as bf16 hs[b,o,t].
// Each wave: 16 M-rows x 64 N-cols (4 wmma accumulators), K-loop 36x32.
// A fetched as f32 (x read once), converted to bf16 in registers.
// B staged once per workgroup into LDS via the Tensor Data Mover.
// ======================================================================
__global__ __launch_bounds__(256)
void conv1_gemm(const float* __restrict__ x, const float* __restrict__ bias,
                const __bf16* __restrict__ Bf, __bf16* __restrict__ hs) {
    __shared__ __bf16 lbf[BF1_ELEM];                   // 147456 B LDS
    int tid  = threadIdx.x;
    int lane = tid & 31;
    int wid  = tid >> 5;

    stage_to_lds(Bf, lbf, BF1_ELEM, tid, wid);

    int wgl  = blockIdx.x * 8 + wid;       // global wave index
    int Mbase = wgl * 16;

    int m  = Mbase + (lane & 15);
    int b_ = m / 42;
    int t_ = m % 42;
    const float* ap = x + (size_t)b_ * 6400 + t_ * 128;
    int khalf = (lane >> 4) << 3;          // 0 or 8

    v8f acc0 = {}; v8f acc1 = {}; v8f acc2 = {}; v8f acc3 = {};
    const v16bf* bfp = (const v16bf*)lbf;

    for (int ks = 0; ks < 36; ++ks) {
        int kb = ks * 32;
        v4f f0 = *(const v4f*)(ap + kb + khalf);
        v4f f1 = *(const v4f*)(ap + kb + khalf + 4);
        v4f f2 = *(const v4f*)(ap + kb + khalf + 16);
        v4f f3 = *(const v4f*)(ap + kb + khalf + 20);
        v16bf a;
#pragma unroll
        for (int j = 0; j < 4; ++j) {
            a[j]      = (__bf16)f0[j];
            a[4 + j]  = (__bf16)f1[j];
            a[8 + j]  = (__bf16)f2[j];
            a[12 + j] = (__bf16)f3[j];
        }
        v16bf bm0 = bfp[(ks * 4 + 0) * 32 + lane];
        v16bf bm1 = bfp[(ks * 4 + 1) * 32 + lane];
        v16bf bm2 = bfp[(ks * 4 + 2) * 32 + lane];
        v16bf bm3 = bfp[(ks * 4 + 3) * 32 + lane];
        acc0 = __builtin_amdgcn_wmma_f32_16x16x32_bf16(false, a, false, bm0, (short)0, acc0, false, false);
        acc1 = __builtin_amdgcn_wmma_f32_16x16x32_bf16(false, a, false, bm1, (short)0, acc1, false, false);
        acc2 = __builtin_amdgcn_wmma_f32_16x16x32_bf16(false, a, false, bm2, (short)0, acc2, false, false);
        acc3 = __builtin_amdgcn_wmma_f32_16x16x32_bf16(false, a, false, bm3, (short)0, acc3, false, false);
    }

    // epilogue: +bias, ReLU, store transposed hs[b, o, t] (bf16)
    int mrb = Mbase + ((lane >> 4) << 3);
    v8f accs[4] = {acc0, acc1, acc2, acc3};
#pragma unroll
    for (int nt = 0; nt < 4; ++nt) {
        int n = nt * 16 + (lane & 15);
        if (n < 50) {
            float bv = bias[n];
#pragma unroll
            for (int r = 0; r < 8; ++r) {
                int mm = mrb + r;
                int bb = mm / 42;
                int tt = mm % 42;
                float v = accs[nt][r] + bv;
                v = v > 0.0f ? v : 0.0f;
                hs[(size_t)bb * HS_PER_B + n * 42 + tt] = (__bf16)v;
            }
        }
    }
}

// ======================================================================
// GEMM2: primary caps conv + bias + SQUASH in-register.
// Each lane's 8 accumulator entries = one squash group (d = s%8).
// ======================================================================
__global__ __launch_bounds__(256)
void pcaps_gemm(const __bf16* __restrict__ hs, const float* __restrict__ bias,
                const __bf16* __restrict__ Bf, float* __restrict__ u) {
    __shared__ __bf16 lbf[BF2_ELEM];                   // 16384 B LDS
    int tid  = threadIdx.x;
    int lane = tid & 31;
    int wid  = tid >> 5;

    stage_to_lds(Bf, lbf, BF2_ELEM, tid, wid);

    int wgl  = blockIdx.x * 8 + wid;
    int Mbase = wgl * 16;                  // 48 % 16 == 0: tile within one b
    int bb    = Mbase / 48;
    int sbase = Mbase % 48;

    int s0 = sbase + (lane & 15);
    int ebase = bb * HS_PER_B + s0 * 42;   // even -> uint aligned
    int khalf = (lane >> 4) << 3;
    const unsigned int* up = (const unsigned int*)hs;

    v8f acc0 = {}; v8f acc1 = {}; v8f acc2 = {}; v8f acc3 = {};
    const v16bf* bfp = (const v16bf*)lbf;

#pragma unroll
    for (int ks = 0; ks < 4; ++ks) {
        int kb = ks * 32;
        int e0 = ebase + kb + khalf;       // even
        int e1 = e0 + 16;
        v8u au;
#pragma unroll
        for (int j = 0; j < 4; ++j) {
            au[j]     = up[(e0 >> 1) + j];
            au[4 + j] = up[(e1 >> 1) + j];
        }
        v16bf a = __builtin_bit_cast(v16bf, au);
        v16bf bm0 = bfp[(ks * 4 + 0) * 32 + lane];
        v16bf bm1 = bfp[(ks * 4 + 1) * 32 + lane];
        v16bf bm2 = bfp[(ks * 4 + 2) * 32 + lane];
        v16bf bm3 = bfp[(ks * 4 + 3) * 32 + lane];
        acc0 = __builtin_amdgcn_wmma_f32_16x16x32_bf16(false, a, false, bm0, (short)0, acc0, false, false);
        acc1 = __builtin_amdgcn_wmma_f32_16x16x32_bf16(false, a, false, bm1, (short)0, acc1, false, false);
        acc2 = __builtin_amdgcn_wmma_f32_16x16x32_bf16(false, a, false, bm2, (short)0, acc2, false, false);
        acc3 = __builtin_amdgcn_wmma_f32_16x16x32_bf16(false, a, false, bm3, (short)0, acc3, false, false);
    }

    int sgrp = sbase + ((lane >> 4) << 3);    // multiple of 8
    v8f accs[4] = {acc0, acc1, acc2, acc3};
#pragma unroll
    for (int nt = 0; nt < 4; ++nt) {
        int oc = nt * 16 + (lane & 15);
        float bv = bias[oc];
        float vals[8];
        float sq = 0.0f;
#pragma unroll
        for (int r = 0; r < 8; ++r) {
            float v = accs[nt][r] + bv;
            vals[r] = v;
            sq += v * v;
        }
        float scale = sq / (1.0f + sq) / (sqrtf(sq) + 1e-8f);
        int cap = oc * 6 + (sgrp >> 3);       // 0..383
        float* uo = u + (size_t)bb * U_PER_B + cap * 8;
#pragma unroll
        for (int r = 0; r < 8; ++r) uo[r] = vals[r] * scale;
    }
}

// ======================================================================
// Routing: one workgroup per batch element, u_hat staged in LDS (f32).
// ======================================================================
__global__ __launch_bounds__(256)
void routing_kernel(const float* __restrict__ u, const float* __restrict__ W,
                    float* __restrict__ out) {
    extern __shared__ float sm[];
    float* tu = sm;             // 5*384*16 = 30720
    float* us = tu + 30720;     // 384*8    =  3072
    float* bl = us + 3072;      // 5*384    =  1920
    float* cl = bl + 1920;      // 5*384    =  1920
    float* sv = cl + 1920;      // 80
    float* vv = sv + 80;        // 80
    int tid = threadIdx.x;
    int b   = blockIdx.x;

    const float* ub = u + (size_t)b * U_PER_B;
    for (int i = tid; i < 3072; i += 256) us[i] = ub[i];
    for (int i = tid; i < 1920; i += 256) bl[i] = 0.0f;
    __syncthreads();

    // u_hat[o,c,d] = sum_i W[o,c,d,i] * u[c,i]
    for (int idx = tid; idx < 30720; idx += 256) {
        int oc = idx >> 4;                   // o*384 + c
        int c  = oc % 384;
        const float* wp = W + (size_t)idx * 8;
        const float* uu = us + c * 8;
        float acc = 0.0f;
#pragma unroll
        for (int i = 0; i < 8; ++i) acc += wp[i] * uu[i];
        tu[idx] = acc;
    }
    __syncthreads();

    for (int iter = 0; iter < 3; ++iter) {
        // softmax over the 5 output caps, per input cap
        for (int c = tid; c < 384; c += 256) {
            float mx = bl[c];
#pragma unroll
            for (int o = 1; o < 5; ++o) mx = fmaxf(mx, bl[o * 384 + c]);
            float e[5], ssum = 0.0f;
#pragma unroll
            for (int o = 0; o < 5; ++o) { e[o] = __expf(bl[o * 384 + c] - mx); ssum += e[o]; }
            float inv = 1.0f / ssum;
#pragma unroll
            for (int o = 0; o < 5; ++o) cl[o * 384 + c] = e[o] * inv;
        }
        __syncthreads();
        // s[o,d] = sum_c c[o,c] * u_hat[o,c,d]
        if (tid < 80) {
            int o = tid >> 4, d = tid & 15;
            const float* tub = tu + (size_t)(o * 384) * 16 + d;
            const float* clb = cl + o * 384;
            float acc = 0.0f;
            for (int c = 0; c < 384; ++c) acc += clb[c] * tub[c * 16];
            sv[tid] = acc;
        }
        __syncthreads();
        // v = squash(s)
        if (tid < 80) {
            int o = tid >> 4;
            float sq = 0.0f;
#pragma unroll
            for (int d = 0; d < 16; ++d) { float t = sv[o * 16 + d]; sq += t * t; }
            float scale = sq / (1.0f + sq) / (sqrtf(sq) + 1e-8f);
            vv[tid] = sv[tid] * scale;
        }
        __syncthreads();
        if (iter < 2) {
            // b[o,c] += dot(u_hat[o,c,:], v[o,:])
            for (int idx = tid; idx < 1920; idx += 256) {
                int o = idx / 384;
                const float* tub = tu + (size_t)idx * 16;
                const float* vb  = vv + o * 16;
                float acc = 0.0f;
#pragma unroll
                for (int d = 0; d < 16; ++d) acc += tub[d] * vb[d];
                bl[idx] += acc;
            }
            __syncthreads();
        }
    }
    if (tid < 80) out[(size_t)b * 80 + tid] = vv[tid];
}

// ======================================================================
extern "C" void kernel_launch(void* const* d_in, const int* in_sizes, int n_in,
                              void* d_out, int out_size, void* d_ws, size_t ws_size,
                              hipStream_t stream) {
    const float* x  = (const float*)d_in[0];   // [4096,50,128]
    const float* w1 = (const float*)d_in[1];   // [50,128,9]
    const float* b1 = (const float*)d_in[2];   // [50]
    const float* pw = (const float*)d_in[3];   // [64,42,3]
    const float* pb = (const float*)d_in[4];   // [64]
    const float* W  = (const float*)d_in[5];   // [5,384,16,8]
    float* out = (float*)d_out;                // [4096,5,16]

    char* ws = (char*)d_ws;
    __bf16* hs  = (__bf16*)(ws + OFF_HS);
    float*  u   = (float*)(ws + OFF_U);
    __bf16* bf1 = (__bf16*)(ws + OFF_BF1);
    __bf16* bf2 = (__bf16*)(ws + OFF_BF2);

    prep_w1<<<BF1_ELEM / 256, 256, 0, stream>>>(w1, bf1);
    prep_w2<<<BF2_ELEM / 256, 256, 0, stream>>>(pw, bf2);
    conv1_gemm<<<L1_M / 128, 256, 0, stream>>>(x, b1, bf1, hs);
    pcaps_gemm<<<L2_M / 128, 256, 0, stream>>>(hs, pb, bf2, u);
    routing_kernel<<<BATCH, 256, 37792 * sizeof(float), stream>>>(u, W, out);
}
